// MambaBlock_7662221656531
// MI455X (gfx1250) — matrix operand order repaired and checked by reading, not compiled, not run on hardware
//
#include <hip/hip_runtime.h>
#include <math.h>

typedef unsigned int   u32;
typedef unsigned short u16;
typedef __attribute__((ext_vector_type(16))) __bf16 v16bf;
typedef __attribute__((ext_vector_type(8)))  float  v8f;
typedef __attribute__((ext_vector_type(8)))  u32    v8u;
typedef __attribute__((ext_vector_type(4)))  u32    v4u;
typedef __attribute__((ext_vector_type(4)))  int    i4v;

#define D_MODEL 1024
#define D_STATE 16
#define D_CONV  4
#define D_INNER 2048
#define BATCH   2
#define SEQLEN  1024

// ---- bf16 round-to-nearest-even helpers -----------------------------------
__device__ __forceinline__ u16 f2bf(float f) {
  u32 u = __float_as_uint(f);
  u += 0x7FFFu + ((u >> 16) & 1u);
  return (u16)(u >> 16);
}
__device__ __forceinline__ float bf2f(u16 h) {
  return __uint_as_float(((u32)h) << 16);
}

// ---- gfx1250 async global->LDS copies (ASYNCcnt), with sync fallback ------
#define AS1 __attribute__((address_space(1)))
#define AS3 __attribute__((address_space(3)))
#if __has_builtin(__builtin_amdgcn_global_load_async_to_lds_b128) && \
    __has_builtin(__builtin_amdgcn_global_load_async_to_lds_b32)  && \
    __has_builtin(__builtin_amdgcn_s_wait_asynccnt)
#define HAVE_ASYNC 1
__device__ __forceinline__ void cp_b128(const u32* g, u32* l) {
  __builtin_amdgcn_global_load_async_to_lds_b128((AS1 i4v*)g, (AS3 i4v*)l, 0, 0);
}
__device__ __forceinline__ void cp_b32(const u32* g, u32* l) {
  __builtin_amdgcn_global_load_async_to_lds_b32((AS1 int*)g, (AS3 int*)l, 0, 0);
}
#define WAIT_ASYNC_N(n) __builtin_amdgcn_s_wait_asynccnt(n)
#else
__device__ __forceinline__ void cp_b128(const u32* g, u32* l) {
  *(v4u*)l = *(const v4u*)g;
}
__device__ __forceinline__ void cp_b32(const u32* g, u32* l) { *l = *g; }
#define WAIT_ASYNC_N(n)
#endif

// ---------------------------------------------------------------------------
// Split fp32 -> bf16 hi + bf16 lo (A-operand layout: natural [M][K] u16)
// ---------------------------------------------------------------------------
__global__ __launch_bounds__(256)
void split_hi_lo(const float* __restrict__ in, u16* __restrict__ hi,
                 u16* __restrict__ lo, int n) {
  int i = blockIdx.x * blockDim.x + threadIdx.x;
  if (i >= n) return;
  float f = in[i];
  u16 h = f2bf(f);
  hi[i] = h;
  lo[i] = f2bf(f - bf2f(h));
}

// ---------------------------------------------------------------------------
// Split fp32 weights -> K-pair-packed bf16 hi/lo:  out[k/2][n] dword =
// pack(bf16(W[2k2][n]), bf16(W[2k2+1][n])).  Makes GEMM B staging a pure copy.
// ---------------------------------------------------------------------------
__global__ __launch_bounds__(256)
void split_pack_w(const float* __restrict__ in, u32* __restrict__ hip,
                  u32* __restrict__ lop, int K2, int N) {
  int i = blockIdx.x * blockDim.x + threadIdx.x;   // over K2*N
  if (i >= K2 * N) return;
  int n  = i % N;
  int k2 = i / N;
  float f0 = in[(size_t)(2 * k2) * N + n];
  float f1 = in[(size_t)(2 * k2 + 1) * N + n];
  u16 h0 = f2bf(f0), h1 = f2bf(f1);
  u16 l0 = f2bf(f0 - bf2f(h0)), l1 = f2bf(f1 - bf2f(h1));
  hip[i] = (u32)h0 | ((u32)h1 << 16);
  lop[i] = (u32)l0 | ((u32)l1 << 16);
}

// ---------------------------------------------------------------------------
// GEMM via bf16x3 split product on V_WMMA_F32_16X16X32_BF16, double-buffered
// LDS with async global->LDS copies.
//   C[M,N] = A[M,K]*W[K,N] + bias,  acc += Ah*Wh + Ah*Wl + Al*Wh (fp32 acc)
// LDS layouts (dwords, 16-byte aligned quads):
//   A: [row 0..63][pair p 0..15]      stride SA=20; frag = quads at p=8hf,+4
//   B: frag-major [n 0..63][pr 0..15] stride SBT=20; frag = quads at pr=8hf,+4
// ---------------------------------------------------------------------------
#define TM 64
#define TN 64
#define TK 32
#define SA  20
#define SBT 20

__device__ __forceinline__ v8f wmma_bf16(v16bf a, v16bf b, v8f c) {
  return __builtin_amdgcn_wmma_f32_16x16x32_bf16(false, a, false, b,
                                                 (short)0, c, false, false);
}

union U8 { v8u u; v4u q[2]; };

__device__ __forceinline__ void stage_tile(const u32* __restrict__ Ah32,
                                           const u32* __restrict__ Al32,
                                           const u32* __restrict__ Whp,
                                           const u32* __restrict__ Wlp,
                                           u32* dAh, u32* dAl, u32* dBh, u32* dBl,
                                           int tid, int m0, int n0, int N, int K,
                                           int k0, bool edge) {
  // A tile: 64 rows x 16 packed pairs; one b128 per thread per half
  {
    int row = tid >> 2;
    int p4  = (tid & 3) << 2;
    size_t gi = (((size_t)(m0 + row) * K + k0) >> 1) + p4;
    cp_b128(Ah32 + gi, dAh + row * SA + p4);
    cp_b128(Al32 + gi, dAl + row * SA + p4);
  }
  // B tile: 16 pair-rows x 64 cols, scattered into frag-major [n][pr]
  if (!edge) {
    #pragma unroll
    for (int i = 0; i < 4; ++i) {
      int v = tid + (i << 8);
      int n = v & 63;
      int r = v >> 6;
      size_t gi = (size_t)((k0 >> 1) + r) * N + (n0 + n);
      cp_b32(Whp + gi, dBh + n * SBT + r);
      cp_b32(Wlp + gi, dBl + n * SBT + r);
    }
  } else {
    #pragma unroll
    for (int i = 0; i < 4; ++i) {
      int v  = tid + (i << 8);
      int n  = v & 63;
      int r  = v >> 6;
      int gc = n0 + n;
      u32 h = 0, l = 0;
      if (gc < N) {
        size_t gi = (size_t)((k0 >> 1) + r) * N + gc;
        h = Whp[gi];
        l = Wlp[gi];
      }
      dBh[n * SBT + r] = h;
      dBl[n * SBT + r] = l;
    }
  }
}

__global__ __launch_bounds__(256)
void gemm_bf16x3(const u16* __restrict__ Ah, const u16* __restrict__ Al,
                 const u32* __restrict__ Whp, const u32* __restrict__ Wlp,
                 const float* __restrict__ bias, float* __restrict__ C,
                 int M, int N, int K) {
  __shared__ alignas(16) u32 sA[2][2][TM * SA];
  __shared__ alignas(16) u32 sB[2][2][TN * SBT];

  const int tid  = threadIdx.x;
  const int wid  = tid >> 5;
  const int lane = tid & 31;
  const int hf   = lane >> 4;
  const int l16  = lane & 15;
  const int waveM = wid & 3;
  const int waveN = wid >> 2;
  const int m0 = blockIdx.y * TM;
  const int n0 = blockIdx.x * TN;
  const bool edge = (n0 + TN > N);

  v8f acc0, acc1;
  #pragma unroll
  for (int i = 0; i < 8; ++i) { acc0[i] = 0.0f; acc1[i] = 0.0f; }

  const u32* Ah32 = (const u32*)Ah;
  const u32* Al32 = (const u32*)Al;
  const int nK = K / TK;

  stage_tile(Ah32, Al32, Whp, Wlp, &sA[0][0][0], &sA[0][1][0],
             &sB[0][0][0], &sB[0][1][0], tid, m0, n0, N, K, 0, edge);

  for (int kt = 0; kt < nK; ++kt) {
    const int cur = kt & 1;
    if (kt + 1 < nK) {
      const int nxt = 1 - cur;
      stage_tile(Ah32, Al32, Whp, Wlp, &sA[nxt][0][0], &sA[nxt][1][0],
                 &sB[nxt][0][0], &sB[nxt][1][0], tid, m0, n0, N, K,
                 (kt + 1) * TK, edge);
      if (edge) { WAIT_ASYNC_N(2); } else { WAIT_ASYNC_N(10); }
    } else {
      WAIT_ASYNC_N(0);
    }
    __syncthreads();

    // ---- fragments: each is two contiguous ds_load_b128 quads ----
    const u32* pAh = &sA[cur][0][0];
    const u32* pAl = &sA[cur][1][0];
    const u32* pBh = &sB[cur][0][0];
    const u32* pBl = &sB[cur][1][0];

    const int abase = (waveM * 16 + l16) * SA + 8 * hf;
    U8 Uah, Ual;
    Uah.q[0] = *(const v4u*)(pAh + abase);
    Uah.q[1] = *(const v4u*)(pAh + abase + 4);
    Ual.q[0] = *(const v4u*)(pAl + abase);
    Ual.q[1] = *(const v4u*)(pAl + abase + 4);
    v16bf ahi = __builtin_bit_cast(v16bf, Uah.u);
    v16bf alo = __builtin_bit_cast(v16bf, Ual.u);

    const int b0 = (waveN * 32 + l16) * SBT + 8 * hf;
    const int b1 = b0 + 16 * SBT;
    U8 Ubh0, Ubl0, Ubh1, Ubl1;
    Ubh0.q[0] = *(const v4u*)(pBh + b0);
    Ubh0.q[1] = *(const v4u*)(pBh + b0 + 4);
    Ubl0.q[0] = *(const v4u*)(pBl + b0);
    Ubl0.q[1] = *(const v4u*)(pBl + b0 + 4);
    Ubh1.q[0] = *(const v4u*)(pBh + b1);
    Ubh1.q[1] = *(const v4u*)(pBh + b1 + 4);
    Ubl1.q[0] = *(const v4u*)(pBl + b1);
    Ubl1.q[1] = *(const v4u*)(pBl + b1 + 4);
    v16bf bh0 = __builtin_bit_cast(v16bf, Ubh0.u);
    v16bf bl0 = __builtin_bit_cast(v16bf, Ubl0.u);
    v16bf bh1 = __builtin_bit_cast(v16bf, Ubh1.u);
    v16bf bl1 = __builtin_bit_cast(v16bf, Ubl1.u);

    // ---- bf16x3 split product, fp32 accumulate ----
    acc0 = wmma_bf16(ahi, bh0, acc0);
    acc0 = wmma_bf16(ahi, bl0, acc0);
    acc0 = wmma_bf16(alo, bh0, acc0);
    acc1 = wmma_bf16(ahi, bh1, acc1);
    acc1 = wmma_bf16(ahi, bl1, acc1);
    acc1 = wmma_bf16(alo, bh1, acc1);
    __syncthreads();
  }

  // ---- store: VGPR r -> row r (lanes 0-15) / r+8 (lanes 16-31) ----
  #pragma unroll
  for (int r = 0; r < 8; ++r) {
    int gr  = m0 + waveM * 16 + r + 8 * hf;
    int gc0 = n0 + waveN * 32 + l16;
    int gc1 = gc0 + 16;
    if (gc0 < N) C[(size_t)gr * N + gc0] = acc0[r] + bias[gc0];
    if (gc1 < N) C[(size_t)gr * N + gc1] = acc1[r] + bias[gc1];
  }
}

// ---------------------------------------------------------------------------
// Causal depthwise conv1d (k=4) + SiLU; emits fp32 xs AND bf16 hi/lo copies.
// ---------------------------------------------------------------------------
__global__ __launch_bounds__(256)
void conv_silu(const float* __restrict__ xz, const float* __restrict__ cw,
               const float* __restrict__ cb, float* __restrict__ xs,
               u16* __restrict__ xsh, u16* __restrict__ xsl) {
  int i = blockIdx.x * blockDim.x + threadIdx.x;   // over B*L*D_INNER
  int d = i & (D_INNER - 1);
  int t = (i >> 11) & (SEQLEN - 1);
  int b = i >> 21;
  float acc = cb[d];
  #pragma unroll
  for (int j = 0; j < D_CONV; ++j) {
    int tl = t + j - (D_CONV - 1);
    if (tl >= 0)
      acc += cw[d * D_CONV + j] *
             xz[(size_t)(b * SEQLEN + tl) * (2 * D_INNER) + d];
  }
  float v = acc / (1.0f + expf(-acc));   // silu
  xs[i] = v;
  u16 h = f2bf(v);
  xsh[i] = h;
  xsl[i] = f2bf(v - bf2f(h));
}

// ---------------------------------------------------------------------------
// dt = softplus(proj[..., :16] @ dt_proj_w + dt_proj_b)   (K=16: VALU)
// ---------------------------------------------------------------------------
__global__ __launch_bounds__(256)
void dtproj_softplus(const float* __restrict__ proj, const float* __restrict__ w,
                     const float* __restrict__ bias, float* __restrict__ dt) {
  int i  = blockIdx.x * blockDim.x + threadIdx.x;  // over B*L*D_INNER
  int d  = i & (D_INNER - 1);
  int bl = i >> 11;
  float acc = bias[d];
  const float* p = proj + (size_t)bl * (2 * D_STATE + D_INNER);
  #pragma unroll
  for (int s = 0; s < D_STATE; ++s) acc += p[s] * w[s * D_INNER + d];
  dt[i] = (acc > 20.0f) ? acc : log1pf(expf(acc));
}

// ---------------------------------------------------------------------------
// Selective scan (reference log-cumsum form) + D skip + z gate.
// One wave = 2 channels; lanes 0-15 states of d0, lanes 16-31 states of d1.
// ---------------------------------------------------------------------------
__global__ __launch_bounds__(256)
void selective_scan(const float* __restrict__ proj, const float* __restrict__ dtf,
                    const float* __restrict__ xs, const float* __restrict__ xz,
                    const float* __restrict__ A_log, const float* __restrict__ Dvec,
                    u16* __restrict__ ymh, u16* __restrict__ yml) {
  const int gtid = blockIdx.x * blockDim.x + threadIdx.x;
  const int wave = gtid >> 5;
  const int lane = threadIdx.x & 31;
  const int s    = lane & 15;
  const int b    = wave / (D_INNER / 2);
  const int d    = (wave % (D_INNER / 2)) * 2 + (lane >> 4);
  const float a  = -expf(A_log[d * D_STATE + s]);
  const float Dd = Dvec[d];
  const int PW   = 2 * D_STATE + D_INNER;   // 2080

  float P = 0.0f, H = 0.0f;
  for (int l = 0; l < SEQLEN; ++l) {
    const size_t row = (size_t)(b * SEQLEN + l);
    const float dt = dtf[row * D_INNER + d];
    const float bp = proj[row * PW + D_STATE + s];
    const float xv = xs[row * D_INNER + d];
    H += expf(P) * (dt * bp * xv);           // uses prefix_{l-1}
    P += logf(expf(dt * a) + 1e-8f);
    float hs = H;
    hs += __shfl_xor(hs, 1);
    hs += __shfl_xor(hs, 2);
    hs += __shfl_xor(hs, 4);
    hs += __shfl_xor(hs, 8);
    if (s == 0) {
      const float cp = proj[row * PW + 2 * D_STATE + d];
      const float zv = xz[row * (2 * D_INNER) + D_INNER + d];
      const float y  = cp * hs + xv * Dd;
      const float o  = y * (zv / (1.0f + expf(-zv)));
      u16 h = f2bf(o);
      ymh[row * D_INNER + d] = h;
      yml[row * D_INNER + d] = f2bf(o - bf2f(h));
    }
  }
}

// ---------------------------------------------------------------------------
extern "C" void kernel_launch(void* const* d_in, const int* in_sizes, int n_in,
                              void* d_out, int out_size, void* d_ws, size_t ws_size,
                              hipStream_t stream) {
  const float* x     = (const float*)d_in[0];
  const float* inW   = (const float*)d_in[1];
  const float* inB   = (const float*)d_in[2];
  const float* convW = (const float*)d_in[3];
  const float* convB = (const float*)d_in[4];
  const float* xpW   = (const float*)d_in[5];
  const float* xpB   = (const float*)d_in[6];
  const float* dtW   = (const float*)d_in[7];
  const float* dtB   = (const float*)d_in[8];
  const float* Alog  = (const float*)d_in[9];
  const float* Dv    = (const float*)d_in[10];
  const float* outW  = (const float*)d_in[11];
  const float* outB  = (const float*)d_in[12];
  float* out = (float*)d_out;
  (void)in_sizes; (void)n_in; (void)out_size; (void)ws_size;

  const int M  = BATCH * SEQLEN;                // 2048
  const int PW = 2 * D_STATE + D_INNER;         // 2080
  const int nX   = M * D_MODEL;
  const int nW1  = D_MODEL * 2 * D_INNER;
  const int nW2  = D_INNER * PW;
  const int nW3  = D_INNER * D_MODEL;
  const int nInn = M * D_INNER;

  char* ws = (char*)d_ws;
  size_t off = 0;
  auto alloc = [&](size_t bytes) {
    void* p = ws + off;
    off += (bytes + 255) & ~(size_t)255;
    return p;
  };
  float* xz   = (float*)alloc((size_t)M * 2 * D_INNER * 4);
  float* proj = (float*)alloc((size_t)M * PW * 4);
  float* dtf  = (float*)alloc((size_t)nInn * 4);
  float* xs   = (float*)alloc((size_t)nInn * 4);
  u16* xh  = (u16*)alloc((size_t)nX * 2);
  u16* xl  = (u16*)alloc((size_t)nX * 2);
  u32* w1h = (u32*)alloc((size_t)nW1 * 2);
  u32* w1l = (u32*)alloc((size_t)nW1 * 2);
  u32* w2h = (u32*)alloc((size_t)nW2 * 2);
  u32* w2l = (u32*)alloc((size_t)nW2 * 2);
  u32* w3h = (u32*)alloc((size_t)nW3 * 2);
  u32* w3l = (u32*)alloc((size_t)nW3 * 2);
  u16* xsh = (u16*)alloc((size_t)nInn * 2);
  u16* xsl = (u16*)alloc((size_t)nInn * 2);
  u16* ymh = (u16*)alloc((size_t)nInn * 2);
  u16* yml = (u16*)alloc((size_t)nInn * 2);

  // 0) one-time hi/lo splits: x (A layout), weights (K-pair-packed B layout)
  split_hi_lo<<<(nX + 255) / 256, 256, 0, stream>>>(x, xh, xl, nX);
  split_pack_w<<<(nW1 / 2 + 255) / 256, 256, 0, stream>>>(inW,  w1h, w1l,
                                                          D_MODEL / 2, 2 * D_INNER);
  split_pack_w<<<(nW2 / 2 + 255) / 256, 256, 0, stream>>>(xpW,  w2h, w2l,
                                                          D_INNER / 2, PW);
  split_pack_w<<<(nW3 / 2 + 255) / 256, 256, 0, stream>>>(outW, w3h, w3l,
                                                          D_INNER / 2, D_MODEL);

  // 1) xz = x @ in_proj_w + b          (2048,1024) x (1024,4096)
  gemm_bf16x3<<<dim3((2 * D_INNER) / TN, M / TM), 256, 0, stream>>>(
      xh, xl, w1h, w1l, inB, xz, M, 2 * D_INNER, D_MODEL);
  // 2) xs = silu(causal_conv(xc))  (+ bf16 hi/lo)
  conv_silu<<<nInn / 256, 256, 0, stream>>>(xz, convW, convB, xs, xsh, xsl);
  // 3) proj = xs @ x_proj_w + b        (2048,2048) x (2048,2080)
  gemm_bf16x3<<<dim3((PW + TN - 1) / TN, M / TM), 256, 0, stream>>>(
      xsh, xsl, w2h, w2l, xpB, proj, M, PW, D_INNER);
  // 4) dt = softplus(proj[:, :16] @ dt_proj_w + b)
  dtproj_softplus<<<nInn / 256, 256, 0, stream>>>(proj, dtW, dtB, dtf);
  // 5) selective scan + D skip + z gate -> ym (bf16 hi/lo)
  selective_scan<<<(BATCH * (D_INNER / 2) * 32) / 256, 256, 0, stream>>>(
      proj, dtf, xs, xz, Alog, Dv, ymh, yml);
  // 6) out = ym @ out_proj_w + b       (2048,2048) x (2048,1024)
  gemm_bf16x3<<<dim3(D_MODEL / TN, M / TM), 256, 0, stream>>>(
      ymh, yml, w3h, w3l, outB, out, M, D_MODEL, D_INNER);
}
